// MultiheadAttention_34626026340738
// MI455X (gfx1250) — matrix operand order repaired and checked by reading, not compiled, not run on hardware
//
#include <hip/hip_runtime.h>
#include <hip/hip_bf16.h>

// ---------------------------------------------------------------------------
// MultiheadAttention forward for MI455X (gfx1250, wave32, WMMA f16->f32).
// B=4, S=2048, D=1024, H=16, Dh=64.
// Pass 0: elementwise f32->f16 convert of x (q,k,v) and W (Wq,Wk,Wv).
// Pass 1: QKV projection GEMM, pure-register WMMA (no LDS), V stored
//         transposed [B,H,Dh,S] so PV B-fragments are contiguous.
// Pass 2: flash attention, 32 query rows per wave, kv tiles of 64.
// Workspace: 48 MB x16 + 6 MB W16 + 48 MB qkv16 = 102 MB in d_ws.
// ---------------------------------------------------------------------------

typedef __attribute__((ext_vector_type(16))) _Float16 v16h;
typedef __attribute__((ext_vector_type(8)))  _Float16 v8h;
typedef __attribute__((ext_vector_type(4)))  _Float16 v4h;
typedef __attribute__((ext_vector_type(8)))  float    v8f;

#define BATCH 4
#define SEQ   2048
#define EMB   1024
#define HEADS 16
#define DH    64

__device__ __forceinline__ v16h combine16(v8h lo, v8h hi) {
    v16h r;
#pragma unroll
    for (int i = 0; i < 8; ++i) { r[i] = lo[i]; r[i + 8] = hi[i]; }
    return r;
}

// B fragment: per-lane 16 contiguous f16 (lane = N, K contiguous)
__device__ __forceinline__ v16h load_b16(const _Float16* p) {
    const v8h* q = (const v8h*)p;
    return combine16(q[0], q[1]);
}

// A fragment: per-lane 16 contiguous f16 at 2*khalf (K contiguous)
__device__ __forceinline__ v16h load_a16(const _Float16* row, int khalf) {
    return combine16(*(const v8h*)(row + 2 * khalf),
                     *(const v8h*)(row + 2 * khalf + 8));
}

__device__ __forceinline__ v8f wmma_f16(v16h a, v16h b, v8f c) {
    return __builtin_amdgcn_wmma_f32_16x16x32_f16(
        false, a, false, b, (short)0, c, false, false);
}

// ---------------------------------------------------------------------------
// Pass 0: f32 -> f16 elementwise (n divisible by 1024). 4 elems/thread.
// ---------------------------------------------------------------------------
__global__ __launch_bounds__(256) void cvt_f32_to_f16_kernel(
    const float* __restrict__ src, _Float16* __restrict__ dst) {
    size_t i = ((size_t)blockIdx.x * 256 + threadIdx.x) * 4;
    float4 f = *(const float4*)(src + i);
    v4h h;
    h[0] = (_Float16)f.x; h[1] = (_Float16)f.y;
    h[2] = (_Float16)f.z; h[3] = (_Float16)f.w;
    *(v4h*)(dst + i) = h;
}

// ---------------------------------------------------------------------------
// Pass 1: out = x @ W.T in f16 (f32 accumulate). No LDS, no barriers.
// grid = ((B*S)/128, HEADS, 3), block = 128 (4 waves, 32 rows each).
// z==2 (V): store transposed [B,H,Dh,S]; else [B,H,S,Dh].
// ---------------------------------------------------------------------------
__global__ __launch_bounds__(128) void qkv_proj_kernel(
    const _Float16* __restrict__ x16, const _Float16* __restrict__ w16,
    _Float16* __restrict__ o16) {
    const int z = blockIdx.z;
    const _Float16* x = x16 + (size_t)z * ((size_t)BATCH * SEQ * EMB);
    const _Float16* W = w16 + (size_t)z * ((size_t)EMB * EMB);
    _Float16*     out = o16 + (size_t)z * ((size_t)BATCH * SEQ * EMB);

    const int h    = blockIdx.y;                       // head = 64-wide N tile
    const int lane = threadIdx.x & 31;
    const int wave = threadIdx.x >> 5;
    const int m0   = blockIdx.x * 128 + wave * 32;     // wave's 32 rows
    const int lrow = lane & 15;
    const int hi   = lane >> 4;
    const int khalf = hi * 8;    // A-fragment half-K base
    const int ks16  = hi * 16;   // B-fragment K base

    const _Float16* xr0 = x + (size_t)(m0 + lrow) * EMB;        // row-tile 0
    const _Float16* xr1 = x + (size_t)(m0 + 16 + lrow) * EMB;   // row-tile 1

    v8f acc[2][4] = {};
    for (int k0 = 0; k0 < EMB; k0 += 32) {
        v16h a0 = load_a16(xr0 + k0, khalf);
        v16h a1 = load_a16(xr1 + k0, khalf);
#pragma unroll
        for (int t = 0; t < 4; ++t) {
            v16h b = load_b16(W + (size_t)(h * 64 + t * 16 + lrow) * EMB +
                              k0 + ks16);
            acc[0][t] = wmma_f16(a0, b, acc[0][t]);
            acc[1][t] = wmma_f16(a1, b, acc[1][t]);
        }
    }

    // D layout: vgpr r <-> row r + 8*hi, lane <-> col lrow
#pragma unroll
    for (int rt = 0; rt < 2; ++rt) {
#pragma unroll
        for (int t = 0; t < 4; ++t) {
#pragma unroll
            for (int r = 0; r < 8; ++r) {
                int row = m0 + rt * 16 + r + 8 * hi;   // row in [B*S]
                int b   = row >> 11;                   // /SEQ
                int s   = row & (SEQ - 1);
                int dh  = t * 16 + lrow;
                _Float16 val = (_Float16)acc[rt][t][r];
                if (z == 2) {  // V transposed: [B,H,Dh,S]
                    out[(((size_t)(b * HEADS + h)) * DH + dh) * SEQ + s] = val;
                } else {       // Q,K: [B,H,S,Dh]
                    out[(((size_t)(b * HEADS + h)) * SEQ + s) * DH + dh] = val;
                }
            }
        }
    }
}

// ---------------------------------------------------------------------------
// Pass 2: flash attention. grid = (S/128, B*H), block = 128 (4 waves).
// Each wave: 32 query rows (2 row-tiles), kv tiles of 64 (4 col-tiles).
// K/V fragments shared across both row-tiles; 32 WMMAs per loop body.
// ---------------------------------------------------------------------------
#define PSTR 72  // P-buffer row stride (f16): 144B rows, 16B-aligned

__global__ __launch_bounds__(128) void attn_kernel(
    const _Float16* __restrict__ q16, const _Float16* __restrict__ k16,
    const _Float16* __restrict__ vT, float* __restrict__ out) {
    __shared__ __align__(16) _Float16 Pld[4][32 * PSTR];

    const int lane = threadIdx.x & 31;
    const int wave = threadIdx.x >> 5;
    const int lrow = lane & 15;
    const int hi   = lane >> 4;
    const int khalf = hi * 8;
    const int ks16  = hi * 16;

    const int bh    = blockIdx.y;                        // b*HEADS + h
    const int qBase = blockIdx.x * 128 + wave * 32;      // query row base

    const _Float16* qh = q16 + (size_t)bh * SEQ * DH;
    const _Float16* kh = k16 + (size_t)bh * SEQ * DH;
    const _Float16* vh = vT  + (size_t)bh * DH * SEQ;

    // Q A-fragments, pre-scaled by 1/sqrt(Dh) = 0.125
    v16h aq[2][2];
#pragma unroll
    for (int rt = 0; rt < 2; ++rt) {
        const _Float16* p = qh + (size_t)(qBase + rt * 16 + lrow) * DH;
#pragma unroll
        for (int f = 0; f < 2; ++f) {
            v16h vq = load_a16(p + f * 32, khalf);
#pragma unroll
            for (int i = 0; i < 16; ++i) vq[i] = vq[i] * (_Float16)0.125f;
            aq[rt][f] = vq;
        }
    }

    v8f acc[2][4] = {};
    float mrow[2][8], lsum[2][8];
#pragma unroll
    for (int rt = 0; rt < 2; ++rt)
#pragma unroll
        for (int r = 0; r < 8; ++r) { mrow[rt][r] = -3.0e38f; lsum[rt][r] = 0.f; }

    _Float16* pbuf = Pld[wave];

    for (int j0 = 0; j0 < SEQ; j0 += 64) {
        // ---- logits: 2 row-tiles x 4 col-tiles, K=Dh=64 ----
        v8f lg[2][4];
#pragma unroll
        for (int c = 0; c < 4; ++c) {
            const _Float16* kp = kh + (size_t)(j0 + c * 16 + lrow) * DH;
            v16h b0 = load_b16(kp + ks16);
            v16h b1 = load_b16(kp + 32 + ks16);
#pragma unroll
            for (int rt = 0; rt < 2; ++rt) {
                v8f zacc = {};
                zacc = wmma_f16(aq[rt][0], b0, zacc);
                zacc = wmma_f16(aq[rt][1], b1, zacc);
                lg[rt][c] = zacc;
            }
        }
        // ---- online softmax (row stats reduced across 16-lane halves) ----
#pragma unroll
        for (int rt = 0; rt < 2; ++rt) {
#pragma unroll
            for (int r = 0; r < 8; ++r) {
                float mx = fmaxf(fmaxf(lg[rt][0][r], lg[rt][1][r]),
                                 fmaxf(lg[rt][2][r], lg[rt][3][r]));
                mx = fmaxf(mx, __shfl_xor(mx, 1, 32));
                mx = fmaxf(mx, __shfl_xor(mx, 2, 32));
                mx = fmaxf(mx, __shfl_xor(mx, 4, 32));
                mx = fmaxf(mx, __shfl_xor(mx, 8, 32));
                float mnew  = fmaxf(mrow[rt][r], mx);
                float alpha = __expf(mrow[rt][r] - mnew);
                float p0 = __expf(lg[rt][0][r] - mnew);
                float p1 = __expf(lg[rt][1][r] - mnew);
                float p2 = __expf(lg[rt][2][r] - mnew);
                float p3 = __expf(lg[rt][3][r] - mnew);
                float ps = (p0 + p1) + (p2 + p3);
                ps += __shfl_xor(ps, 1, 32);
                ps += __shfl_xor(ps, 2, 32);
                ps += __shfl_xor(ps, 4, 32);
                ps += __shfl_xor(ps, 8, 32);
                lsum[rt][r] = lsum[rt][r] * alpha + ps;
                mrow[rt][r] = mnew;
#pragma unroll
                for (int t = 0; t < 4; ++t) acc[rt][t][r] *= alpha;
                int m = rt * 16 + r + 8 * hi;  // C-layout row in P buffer
                pbuf[m * PSTR + lrow]      = (_Float16)p0;
                pbuf[m * PSTR + 16 + lrow] = (_Float16)p1;
                pbuf[m * PSTR + 32 + lrow] = (_Float16)p2;
                pbuf[m * PSTR + 48 + lrow] = (_Float16)p3;
            }
        }
        // DS ops are in-order per wave; make the store->load dep explicit
        asm volatile("s_wait_dscnt 0" ::: "memory");
        // ---- P (2x 16x64) as A-fragments, V^T rows as B-fragments ----
        v16h ap[2][2];
#pragma unroll
        for (int rt = 0; rt < 2; ++rt) {
            const _Float16* pr = pbuf + (rt * 16 + lrow) * PSTR;
            ap[rt][0] = load_a16(pr, khalf);
            ap[rt][1] = load_a16(pr + 32, khalf);
        }
#pragma unroll
        for (int t = 0; t < 4; ++t) {
            const _Float16* vp = vh + (size_t)(t * 16 + lrow) * SEQ + j0;
            v16h bv0 = load_b16(vp + ks16);
            v16h bv1 = load_b16(vp + 32 + ks16);
#pragma unroll
            for (int rt = 0; rt < 2; ++rt) {
                acc[rt][t] = wmma_f16(ap[rt][0], bv0, acc[rt][t]);
                acc[rt][t] = wmma_f16(ap[rt][1], bv1, acc[rt][t]);
            }
        }
    }

    // ---- normalize and store f32 output [B,S,H*Dh] ----
    const int b = bh >> 4;          // /HEADS
    const int h = bh & (HEADS - 1);
#pragma unroll
    for (int rt = 0; rt < 2; ++rt) {
#pragma unroll
        for (int r = 0; r < 8; ++r) {
            float inv = 1.0f / lsum[rt][r];
            int s = qBase + rt * 16 + r + 8 * hi;
#pragma unroll
            for (int t = 0; t < 4; ++t) {
                out[((size_t)b * SEQ + s) * EMB + h * DH + t * 16 + lrow] =
                    acc[rt][t][r] * inv;
            }
        }
    }
}

// ---------------------------------------------------------------------------
extern "C" void kernel_launch(void* const* d_in, const int* in_sizes, int n_in,
                              void* d_out, int out_size, void* d_ws,
                              size_t ws_size, hipStream_t stream) {
    const float* xin[3] = {(const float*)d_in[0], (const float*)d_in[1],
                           (const float*)d_in[2]};
    const float* win[3] = {(const float*)d_in[3], (const float*)d_in[4],
                           (const float*)d_in[5]};
    float* out = (float*)d_out;

    const size_t xe = (size_t)BATCH * SEQ * EMB;  // 8M elems per input
    const size_t we = (size_t)EMB * EMB;          // 1M elems per weight
    _Float16* x16 = (_Float16*)d_ws;              // 3*xe halves (48 MB)
    _Float16* w16 = x16 + 3 * xe;                 // 3*we halves (6 MB)
    _Float16* o16 = w16 + 3 * we;                 // 3*xe halves (48 MB)

    // Pass 0: downconvert activations and weights
    for (int z = 0; z < 3; ++z) {
        cvt_f32_to_f16_kernel<<<xe / 1024, 256, 0, stream>>>(xin[z],
                                                             x16 + z * xe);
        cvt_f32_to_f16_kernel<<<we / 1024, 256, 0, stream>>>(win[z],
                                                             w16 + z * we);
    }
    // Pass 1: projections (Q,K -> [B,H,S,Dh], V -> [B,H,Dh,S])
    qkv_proj_kernel<<<dim3((BATCH * SEQ) / 128, HEADS, 3), 128, 0, stream>>>(
        x16, w16, o16);
    // Pass 2: flash attention
    attn_kernel<<<dim3(SEQ / 128, BATCH * HEADS), 128, 0, stream>>>(
        o16, o16 + xe, o16 + 2 * xe, out);
}
